// GSF2AppV2_correlation_26834955665786
// MI455X (gfx1250) — compile-verified, matrix operand order checked by reading, not run on hardware
//
#include <hip/hip_runtime.h>
#include <hip/hip_bf16.h>

// ---------------------------------------------------------------------------
// MI455X (gfx1250) implementation, round 3.
// Roofline: single-use sfm_fw (419 MB f32) dominates HBM traffic -> ~18us
// floor at 23.3 TB/s. Activations (<=105 MB) stay resident in the 192 MB L2.
// All GEMM-shaped work uses v_wmma_f32_16x16x32_f16 (f32 accumulate).
// Round-3 changes: (a) K-loop unroll x2 on the vectorized paths so 16 b128
// loads are in flight per iteration (weight-stream saturation), (b)
// global_prefetch_b8 on the split-K weight stream, (c) compile-time RELU to
// drop the epilogue cndmask selects.
// ---------------------------------------------------------------------------

typedef __attribute__((ext_vector_type(16))) _Float16 v16h;
typedef __attribute__((ext_vector_type(8)))  float    v8f;

__device__ __forceinline__ v8f wmma16(v16h a, v16h b, v8f c) {
  return __builtin_amdgcn_wmma_f32_16x16x32_f16(false, a, false, b,
                                                (short)0, c, false, false);
}

// Branch-free f32->f16 fragment pack from four float4 loads.
__device__ __forceinline__ v16h cvt_frag(float4 x0, float4 x1, float4 y0,
                                         float4 y1) {
  v16h f;
  f[0] = (_Float16)x0.x;  f[1] = (_Float16)x0.y;
  f[2] = (_Float16)x0.z;  f[3] = (_Float16)x0.w;
  f[4] = (_Float16)x1.x;  f[5] = (_Float16)x1.y;
  f[6] = (_Float16)x1.z;  f[7] = (_Float16)x1.w;
  f[8] = (_Float16)y0.x;  f[9] = (_Float16)y0.y;
  f[10] = (_Float16)y0.z; f[11] = (_Float16)y0.w;
  f[12] = (_Float16)y1.x; f[13] = (_Float16)y1.y;
  f[14] = (_Float16)y1.z; f[15] = (_Float16)y1.w;
  return f;
}

// Fast fragment load: p = row_base + k0 + kb, 16B-aligned (lda%4==0, K%4==0,
// k0%32==0, kb in {0,8}).  Lane layout per ISA 7.12.2: halves 0..7 = K
// kb+0..7, halves 8..15 = K kb+16..23.
__device__ __forceinline__ v16h load_frag_vec(const float* __restrict__ p) {
  const float4* q = reinterpret_cast<const float4*>(p);
  return cvt_frag(q[0], q[1], q[4], q[5]);
}

// Guarded scalar fragment load, branch-free: unconditional loads from
// clamped addresses, zeroed with selects (v_cndmask), no EXEC games.
__device__ __forceinline__ v16h load_frag_guard(const float* __restrict__ row,
                                                bool ok, int k0, int kb,
                                                int K) {
  v16h f;
#pragma unroll
  for (int h = 0; h < 2; ++h) {
#pragma unroll
    for (int i = 0; i < 8; ++i) {
      const int k = k0 + kb + h * 16 + i;
      const int kc = k < K ? k : K - 1;   // safe address
      float v = row[kc];                  // unconditional load
      v = (ok && k < K) ? v : 0.0f;       // branch-free zeroing
      f[h * 8 + i] = (_Float16)v;
    }
  }
  return f;
}

// ---------------------------------------------------------------------------
// Fast FC: requires M%16==0, N%16==0, K%64==0 here (call sites: 2048, 512,
// 2560), lda%4==0, 16B-aligned bases.  One wave = one 16x16 tile; float4
// fragment loads; K-loop unrolled x2 for deeper vmem pipelining.
// ---------------------------------------------------------------------------
template <bool RELU>
__global__ void wmma_fc_fast_kernel(const float* __restrict__ A, int lda,
                                    const float* __restrict__ W,
                                    const float* __restrict__ bias,
                                    float* __restrict__ C, int ldc,
                                    int M, int N, int K) {
  const int lane = threadIdx.x & 31;
  const int wave = threadIdx.x >> 5;
  const int mTiles = M >> 4;
  const int nTiles = N >> 4;
  const int tile = blockIdx.x * (blockDim.x >> 5) + wave;
  if (tile >= mTiles * nTiles) return;
  const int m0 = (tile % mTiles) << 4;
  const int n0 = (tile / mTiles) << 4;
  const int rc = lane & 15;
  const int kb = (lane < 16) ? 0 : 8;
  const float* Ap = A + (size_t)(m0 + rc) * lda + kb;
  const float* Wp = W + (size_t)(n0 + rc) * K + kb;

  v8f acc = {};
#pragma unroll 2
  for (int k0 = 0; k0 < K; k0 += 32) {
    const v16h a = load_frag_vec(Ap + k0);
    const v16h b = load_frag_vec(Wp + k0);
    acc = wmma16(a, b, acc);
  }

  const int moff = (lane < 16) ? 0 : 8;
  const int cn = n0 + rc;
  const float bv = bias[cn];
#pragma unroll
  for (int v = 0; v < 8; ++v) {
    float val = acc[v] + bv;
    if (RELU) val = fmaxf(val, 0.0f);
    C[(size_t)(m0 + moff + v) * ldc + cn] = val;
  }
}

// ---------------------------------------------------------------------------
// Split-K fast FC for the 419 MB sfm_fw stream: 64 K-chunks -> ~16K waves
// streaming weights concurrently; accumulate onto bias-seeded C with f32
// global atomics (no activation -> order-free).  Prefetch the weight stream
// ~2KB ahead (global_prefetch_b8); K-loop unrolled x2 (50 steps/chunk).
// ---------------------------------------------------------------------------
__global__ void wmma_fc_splitk_kernel(const float* __restrict__ A, int lda,
                                      const float* __restrict__ W,
                                      float* __restrict__ C, int ldc,
                                      int M, int N, int K, int kChunk) {
  const int lane = threadIdx.x & 31;
  const int wave = threadIdx.x >> 5;
  const int mTiles = M >> 4;
  const int nTiles = N >> 4;
  const int chunks = K / kChunk;
  const long long tile = (long long)blockIdx.x * (blockDim.x >> 5) + wave;
  if (tile >= (long long)mTiles * nTiles * chunks) return;
  const int mT = (int)(tile % mTiles);
  const int rest = (int)(tile / mTiles);
  const int nT = rest % nTiles;
  const int ch = rest / nTiles;
  const int m0 = mT << 4, n0 = nT << 4;
  const int kBeg = ch * kChunk;
  const int kEnd = kBeg + kChunk;

  const int rc = lane & 15;
  const int kb = (lane < 16) ? 0 : 8;
  const float* Ap = A + (size_t)(m0 + rc) * lda + kb;
  const float* Wp = W + (size_t)(n0 + rc) * K + kb;

  v8f acc = {};
#pragma unroll 2
  for (int k0 = kBeg; k0 < kEnd; k0 += 32) {
    __builtin_prefetch(Wp + k0 + 512, 0, 1);  // ~2KB ahead on weight stream
    const v16h a = load_frag_vec(Ap + k0);
    const v16h b = load_frag_vec(Wp + k0);
    acc = wmma16(a, b, acc);
  }

  const int moff = (lane < 16) ? 0 : 8;
  const int cn = n0 + rc;
#pragma unroll
  for (int v = 0; v < 8; ++v)
    atomicAdd(&C[(size_t)(m0 + moff + v) * ldc + cn], acc[v]);
}

// ---------------------------------------------------------------------------
// Generic FC (ragged K / N): clamped scalar loads, branch-free selects.
// Used only for the tiny fusion FCs (K=897, N=10).
// ---------------------------------------------------------------------------
template <bool RELU>
__global__ void wmma_fc_gen_kernel(const float* __restrict__ A, int lda,
                                   const float* __restrict__ W,
                                   const float* __restrict__ bias,
                                   float* __restrict__ C, int ldc,
                                   int M, int N, int K) {
  const int lane = threadIdx.x & 31;
  const int wave = threadIdx.x >> 5;
  const int mTiles = (M + 15) >> 4;
  const int nTiles = (N + 15) >> 4;
  const int tile = blockIdx.x * (blockDim.x >> 5) + wave;
  if (tile >= mTiles * nTiles) return;
  const int m0 = (tile % mTiles) << 4;
  const int n0 = (tile / mTiles) << 4;

  const int rc = lane & 15;
  const int kb = (lane < 16) ? 0 : 8;
  const int row = m0 + rc;
  const int col = n0 + rc;
  const bool rowOk = row < M;
  const bool colOk = col < N;
  const float* Arow = A + (size_t)(rowOk ? row : M - 1) * lda;
  const float* Wrow = W + (size_t)(colOk ? col : N - 1) * K;

  v8f acc = {};
  for (int k0 = 0; k0 < K; k0 += 32) {
    const v16h a = load_frag_guard(Arow, rowOk, k0, kb, K);
    const v16h b = load_frag_guard(Wrow, colOk, k0, kb, K);
    acc = wmma16(a, b, acc);
  }

  const int moff = (lane < 16) ? 0 : 8;
  const int cn = n0 + rc;
  if (cn < N) {
    const float bv = bias[cn];
#pragma unroll
    for (int v = 0; v < 8; ++v) {
      const int cm = m0 + moff + v;
      if (cm < M) {
        float val = acc[v] + bv;
        if (RELU) val = fmaxf(val, 0.0f);
        C[(size_t)cm * ldc + cn] = val;
      }
    }
  }
}

// ---------------------------------------------------------------------------
// Implicit-GEMM 3x3 pad-1 conv (NCHW / OIHW).  M = Nb*H*W (mult of 16),
// N = Cout (mult of 16), K = Cin*9.  OIHW weights flatten to the N x K GEMM
// operand.  A-gather is branch-free: clamped indices + select for padding.
// Hot loops (K%32==0 portion) carry no k<K compares; the ragged tail (only
// conv1, K=45) is a single separate step.
// ---------------------------------------------------------------------------
__device__ __forceinline__ float conv_a_elem(const float* __restrict__ Inb,
                                             int k, int oy, int ox, int H,
                                             int Wd, int HW) {
  const int ci = k / 9;
  const int t = k - ci * 9;
  const int ky = t / 3;
  const int kx = t - ky * 3;
  const int iy = oy + ky - 1;
  const int ix = ox + kx - 1;
  const bool in =
      ((unsigned)iy < (unsigned)H) && ((unsigned)ix < (unsigned)Wd);
  const int iyc = in ? iy : 0;
  const int ixc = in ? ix : 0;
  const float v = Inb[(size_t)ci * HW + iyc * Wd + ixc];  // always safe
  return in ? v : 0.0f;
}

__global__ void wmma_conv3x3_relu_kernel(const float* __restrict__ In,
                                         const float* __restrict__ Wt,
                                         const float* __restrict__ bias,
                                         float* __restrict__ Out,
                                         int Nb, int Cin, int H, int Wd,
                                         int Cout) {
  const int lane = threadIdx.x & 31;
  const int wave = threadIdx.x >> 5;
  const int HW = H * Wd;
  const int M = Nb * HW;
  const int K = Cin * 9;
  const int mTiles = M >> 4;
  const int nTiles = Cout >> 4;
  const int tile = blockIdx.x * (blockDim.x >> 5) + wave;
  if (tile >= mTiles * nTiles) return;
  const int m0 = (tile % mTiles) << 4;
  const int n0 = (tile / mTiles) << 4;

  const int rc = lane & 15;
  const int kb = (lane < 16) ? 0 : 8;

  // A-matrix row = output position for this lane.
  const int m = m0 + rc;
  const int n = m / HW;
  const int r = m - n * HW;
  const int oy = r / Wd;
  const int ox = r - oy * Wd;
  const float* Inb = In + (size_t)n * Cin * HW;

  const int co = n0 + rc;
  const float* Wrow = Wt + (size_t)co * K;

  v8f acc = {};
  const int Kmain = K & ~31;
  for (int k0 = 0; k0 < Kmain; k0 += 32) {
    v16h a, b;
#pragma unroll
    for (int h = 0; h < 2; ++h) {
#pragma unroll
      for (int i = 0; i < 8; ++i) {
        const int k = k0 + kb + h * 16 + i;  // always < K here
        a[h * 8 + i] = (_Float16)conv_a_elem(Inb, k, oy, ox, H, Wd, HW);
        b[h * 8 + i] = (_Float16)Wrow[k];
      }
    }
    acc = wmma16(a, b, acc);
  }
  if (Kmain < K) {  // ragged tail (conv1: K=45)
    v16h a, b;
#pragma unroll
    for (int h = 0; h < 2; ++h) {
#pragma unroll
      for (int i = 0; i < 8; ++i) {
        const int k = Kmain + kb + h * 16 + i;
        const int kc = k < K ? k : K - 1;
        const bool ok = k < K;
        float av = conv_a_elem(Inb, kc, oy, ox, H, Wd, HW);
        float bv = Wrow[kc];
        a[h * 8 + i] = (_Float16)(ok ? av : 0.0f);
        b[h * 8 + i] = (_Float16)(ok ? bv : 0.0f);
      }
    }
    acc = wmma16(a, b, acc);
  }

  const int moff = (lane < 16) ? 0 : 8;
  const float bv = bias[co];
#pragma unroll
  for (int v = 0; v < 8; ++v) {
    const int cm = m0 + moff + v;
    const int bn = cm / HW;
    const int rr = cm - bn * HW;
    Out[((size_t)bn * Cout + co) * HW + rr] = fmaxf(acc[v] + bv, 0.0f);
  }
}

// --------------------------- small VALU kernels ----------------------------

__global__ void conv1d3_relu_kernel(const float* __restrict__ in,
                                    const float* __restrict__ w,
                                    const float* __restrict__ b,
                                    float* __restrict__ out,
                                    int Nb, int Cin, int Cout, int L) {
  const int idx = blockIdx.x * blockDim.x + threadIdx.x;
  const int total = Nb * Cout * L;
  if (idx >= total) return;
  const int n = idx / (Cout * L);
  const int r = idx - n * Cout * L;
  const int co = r / L;
  const int x = r - co * L;
  const float* inb = in + (size_t)n * Cin * L;
  const float* wb = w + (size_t)co * Cin * 3;
  float acc = b[co];
  for (int ci = 0; ci < Cin; ++ci) {
#pragma unroll
    for (int t = 0; t < 3; ++t) {
      const int xi = x + t - 1;
      const int xc = (xi < 0) ? 0 : (xi >= L ? L - 1 : xi);
      const float v = inb[ci * L + xc];
      acc += ((xi >= 0) && (xi < L)) ? v * wb[ci * 3 + t] : 0.0f;
    }
  }
  out[idx] = fmaxf(acc, 0.0f);
}

__global__ void maxpool3s2p1_kernel(const float* __restrict__ in,
                                    float* __restrict__ out,
                                    int NC, int H, int Wd, int Ho, int Wo) {
  const int idx = blockIdx.x * blockDim.x + threadIdx.x;
  const int total = NC * Ho * Wo;
  if (idx >= total) return;
  const int c = idx / (Ho * Wo);
  const int r = idx - c * (Ho * Wo);
  const int oy = r / Wo;
  const int ox = r - oy * Wo;
  const float* p = in + (size_t)c * H * Wd;
  float m = -INFINITY;
#pragma unroll
  for (int dy = 0; dy < 3; ++dy) {
    const int iy = oy * 2 - 1 + dy;
#pragma unroll
    for (int dx = 0; dx < 3; ++dx) {
      const int ix = ox * 2 - 1 + dx;
      const bool in_b =
          ((unsigned)iy < (unsigned)H) && ((unsigned)ix < (unsigned)Wd);
      const int iyc = in_b ? iy : 0;
      const int ixc = in_b ? ix : 0;
      const float v = p[iyc * Wd + ixc];
      m = fmaxf(m, in_b ? v : -INFINITY);
    }
  }
  out[idx] = m;
}

// x_fus0[b, z] = sum_{k<256} feats[b, 2z + k] * kern[b, k]  (b<64, z<897)
__global__ void corr_kernel(const float* __restrict__ feats,
                            const float* __restrict__ kern,
                            float* __restrict__ out) {
  const int idx = blockIdx.x * blockDim.x + threadIdx.x;
  if (idx >= 64 * 897) return;
  const int bIdx = idx / 897;
  const int z = idx - bIdx * 897;
  const float* f = feats + (size_t)bIdx * 2048 + 2 * z;
  const float* kr = kern + (size_t)bIdx * 256;
  float s = 0.0f;
#pragma unroll 8
  for (int k = 0; k < 256; ++k) s += f[k] * kr[k];
  out[idx] = s;
}

__global__ void seed_bias_kernel(float* __restrict__ C, int ldc,
                                 const float* __restrict__ bias,
                                 int M, int N) {
  const int idx = blockIdx.x * blockDim.x + threadIdx.x;
  if (idx >= M * N) return;
  const int m = idx / N;
  const int n = idx - m * N;
  C[(size_t)m * ldc + n] = bias[n];
}

// ---------------------------------------------------------------------------

static inline int fc_blocks(int M, int N) {
  const int tiles = ((M + 15) / 16) * ((N + 15) / 16);
  return (tiles + 7) / 8;  // 8 waves per 256-thread block
}

extern "C" void kernel_launch(void* const* d_in, const int* in_sizes, int n_in,
                              void* d_out, int out_size, void* d_ws,
                              size_t ws_size, hipStream_t stream) {
  (void)in_sizes; (void)n_in; (void)out_size; (void)ws_size;
  const float* cnn_feats = (const float*)d_in[0];   // 64 x 2048
  const float* sfv_x     = (const float*)d_in[1];   // 64 x 1 x 80
  const float* sfm_x     = (const float*)d_in[2];   // 64 x 5 x 80 x 80
  const float* cnn_w1    = (const float*)d_in[3];   // 512 x 2048
  const float* cnn_b1    = (const float*)d_in[4];
  const float* cnn_w2    = (const float*)d_in[5];   // 256 x 512
  const float* cnn_b2    = (const float*)d_in[6];
  const float* sfv_cw1   = (const float*)d_in[7];   // 32 x 1 x 3
  const float* sfv_cb1   = (const float*)d_in[8];
  const float* sfv_cw2   = (const float*)d_in[9];   // 128 x 32 x 3
  const float* sfv_cb2   = (const float*)d_in[10];
  const float* sfv_cw3   = (const float*)d_in[11];  // 32 x 128 x 3
  const float* sfv_cb3   = (const float*)d_in[12];
  const float* sfv_fw    = (const float*)d_in[13];  // 1024 x 2560
  const float* sfv_fb    = (const float*)d_in[14];
  const float* sfm_cw1   = (const float*)d_in[15];  // 64 x 5 x 3 x 3
  const float* sfm_cb1   = (const float*)d_in[16];
  const float* sfm_cw2   = (const float*)d_in[17];  // 128 x 64 x 3 x 3
  const float* sfm_cb2   = (const float*)d_in[18];
  const float* sfm_cw3   = (const float*)d_in[19];  // 64 x 128 x 3 x 3
  const float* sfm_cb3   = (const float*)d_in[20];
  const float* sfm_fw    = (const float*)d_in[21];  // 1024 x 102400
  const float* sfm_fb    = (const float*)d_in[22];
  const float* fus_w1    = (const float*)d_in[23];  // 512 x 897
  const float* fus_b1    = (const float*)d_in[24];
  const float* fus_w2    = (const float*)d_in[25];  // 10 x 512
  const float* fus_b2    = (const float*)d_in[26];

  float* ws = (float*)d_ws;
  float* x1    = ws;                      // 64*512
  float* kern  = x1    + 64 * 512;        // 64*256
  float* sa1   = kern  + 64 * 256;        // 64*32*80
  float* sa2   = sa1   + 64 * 32 * 80;    // 64*128*80
  float* sa3   = sa2   + 64 * 128 * 80;   // 64*32*80
  float* feats = sa3   + 64 * 32 * 80;    // 64*2048  (sfv | sfm concat)
  float* ma1   = feats + 64 * 2048;       // 64*64*80*80
  float* mp    = ma1   + (size_t)64 * 64 * 80 * 80;  // 64*64*40*40
  float* ma2   = mp    + (size_t)64 * 64 * 40 * 40;  // 64*128*40*40
  float* ma3   = ma2   + (size_t)64 * 128 * 40 * 40; // 64*64*40*40
  float* cor   = ma3   + (size_t)64 * 64 * 40 * 40;  // 64*897
  float* f1    = cor   + 64 * 897;        // 64*512
  float* outp  = (float*)d_out;           // 64*10

  // ---- CNN MLP branch ("kern") ----
  wmma_fc_fast_kernel<true><<<fc_blocks(64, 512), 256, 0, stream>>>(
      cnn_feats, 2048, cnn_w1, cnn_b1, x1, 512, 64, 512, 2048);
  wmma_fc_fast_kernel<true><<<fc_blocks(64, 256), 256, 0, stream>>>(
      x1, 512, cnn_w2, cnn_b2, kern, 256, 64, 256, 512);

  // ---- SFV branch (tiny 1D convs + FC into feats[:, 0:1024]) ----
  conv1d3_relu_kernel<<<(64 * 32 * 80 + 255) / 256, 256, 0, stream>>>(
      sfv_x, sfv_cw1, sfv_cb1, sa1, 64, 1, 32, 80);
  conv1d3_relu_kernel<<<(64 * 128 * 80 + 255) / 256, 256, 0, stream>>>(
      sa1, sfv_cw2, sfv_cb2, sa2, 64, 32, 128, 80);
  conv1d3_relu_kernel<<<(64 * 32 * 80 + 255) / 256, 256, 0, stream>>>(
      sa2, sfv_cw3, sfv_cb3, sa3, 64, 128, 32, 80);
  wmma_fc_fast_kernel<false><<<fc_blocks(64, 1024), 256, 0, stream>>>(
      sa3, 2560, sfv_fw, sfv_fb, feats, 2048, 64, 1024, 2560);

  // ---- SFM branch (implicit-GEMM WMMA convs) ----
  {
    const int tiles1 = (64 * 80 * 80 / 16) * (64 / 16);
    wmma_conv3x3_relu_kernel<<<(tiles1 + 7) / 8, 256, 0, stream>>>(
        sfm_x, sfm_cw1, sfm_cb1, ma1, 64, 5, 80, 80, 64);
  }
  maxpool3s2p1_kernel<<<(64 * 64 * 40 * 40 + 255) / 256, 256, 0, stream>>>(
      ma1, mp, 64 * 64, 80, 80, 40, 40);
  {
    const int tiles2 = (64 * 40 * 40 / 16) * (128 / 16);
    wmma_conv3x3_relu_kernel<<<(tiles2 + 7) / 8, 256, 0, stream>>>(
        mp, sfm_cw2, sfm_cb2, ma2, 64, 64, 40, 40, 128);
    const int tiles3 = (64 * 40 * 40 / 16) * (64 / 16);
    wmma_conv3x3_relu_kernel<<<(tiles3 + 7) / 8, 256, 0, stream>>>(
        ma2, sfm_cw3, sfm_cb3, ma3, 64, 128, 40, 40, 64);
  }

  // ---- sfm FC into feats[:, 1024:2048]: bias-seed then split-K atomics ----
  seed_bias_kernel<<<(64 * 1024 + 255) / 256, 256, 0, stream>>>(
      feats + 1024, 2048, sfm_fb, 64, 1024);
  {
    const int kChunk = 1600;                       // 64 chunks, 50 steps each
    const int tiles = 4 * 64 * (102400 / kChunk);  // 16384 waves
    wmma_fc_splitk_kernel<<<(tiles + 7) / 8, 256, 0, stream>>>(
        ma3, 102400, sfm_fw, feats + 1024, 2048, 64, 1024, 102400, kChunk);
  }

  // ---- fusion: sliding-window correlation + two FCs ----
  corr_kernel<<<(64 * 897 + 255) / 256, 256, 0, stream>>>(feats, kern, cor);
  wmma_fc_gen_kernel<true><<<fc_blocks(64, 512), 256, 0, stream>>>(
      cor, 897, fus_w1, fus_b1, f1, 512, 64, 512, 897);
  wmma_fc_gen_kernel<false><<<fc_blocks(64, 10), 256, 0, stream>>>(
      f1, 512, fus_w2, fus_b2, outp, 10, 64, 10, 512);
}
